// SelfAttentionV2_86028194938986
// MI455X (gfx1250) — matrix operand order repaired and checked by reading, not compiled, not run on hardware
//
#include <hip/hip_runtime.h>
#include <hip/hip_bf16.h>

// Causal self-attention for MI455X (gfx1250), wave32 + WMMA.
//  N=4096, D_IN=D_OUT=768, f32 in/out. All heavy math on
//  v_wmma_f32_16x16x32_f16 (f16 operands, f32 accumulate).
//
// Pipeline (all scratch in d_ws, ~62 MB; entire working set is L2-resident
// on a 192 MB L2 part, so materializing S/P is essentially free and the
// problem is pure matrix throughput):
//   0) cvt:     x, Wq, Wk, Wv -> f16 copies (removes f32->f16 cvts from all
//               GEMM inner loops)
//   1) qkv:     Q = (x Wq^T) * 1/sqrt(768) -> f16 [4096,768]
//               K =  x Wk^T                -> f16 [4096,768]
//               Vt = (x Wv^T)^T            -> f16 [768,4096]
//   2) scores:  S = Q K^T -> f16 [4096,4096]; blocks above diagonal skipped
//   3) softmax: in-place causal row softmax, zeros for j>i  (S -> P)
//   4) pv:      out = P V (via Vt) -> f32, key blocks <= diagonal only
//
// GEMM core: 8 waves/block, block tile 128x128, wave tile 32x64
// (2 A-strips x 4 B-tiles). Per k-step all 6 B/A fragments are loaded into
// distinct registers first, then 8 WMMAs issue back-to-back so the
// WMMA->VALU operand-overwrite hazard NOPs amortize.

typedef __attribute__((ext_vector_type(16))) _Float16 v16h;
typedef __attribute__((ext_vector_type(8)))  _Float16 v8h;
typedef __attribute__((ext_vector_type(8)))  float    v8f;
typedef __attribute__((ext_vector_type(4)))  float    v4f;

union U16h { v16h v; v8h h[2]; };

#define N_TOK 4096
#define D_DIM 768
#define SOFTMAX_SCALE 0.03608439182435161f  // 1/sqrt(768)

// ---- WMMA fragment helpers (layouts per CDNA5 ISA 7.12.2) ----

// A-matrix 16x32 f16, source row-major. lanes 0-15: row M=lane, K in
// {0..7,16..23}; lanes 16-31: row M=lane-16, K in {8..15,24..31}.
__device__ inline v16h load_a_f16(const _Float16* base, int ld) {
  const int lane = threadIdx.x & 31;
  const int m  = lane & 15;
  const int kq = (lane >> 4) << 3;            // 0 or 8
  const _Float16* p = base + m * ld + kq;
  U16h u;
  u.h[0] = *(const v8h*)(p);
  u.h[1] = *(const v8h*)(p + 16);
  return u.v;
}

// B-matrix 32x16 where B = M^T, M row-major: B[k][n] = M[n][k].
// lanes 0-15: column n=lane, K in {0..7,16..23}; lanes 16-31: n=lane-16,
// K in {8..15,24..31}.
__device__ inline v16h load_bt_f16(const _Float16* base, int ld) {
  const int lane = threadIdx.x & 31;
  const int n  = lane & 15;
  const int kq = (lane >> 4) << 3;            // 0 or 8
  const _Float16* p = base + n * ld + kq;
  U16h u;
  u.h[0] = *(const v8h*)(p);
  u.h[1] = *(const v8h*)(p + 16);
  return u.v;
}

__device__ inline v8f wmma_f16(v16h a, v16h b, v8f c) {
  return __builtin_amdgcn_wmma_f32_16x16x32_f16(
      false, a, false, b, (short)0, c, false, false);
}

// ---- Wave-level 32x64 GEMM core ----------------------------------------
// acc[s*4+t]: strip s (rows +16*s), column tile t (cols +16*t).
// A: wave's first strip row base; B: wave's 64-col group base row in the
// transposed-B source (row-major, K contiguous).
__device__ inline void gemm32x64(const _Float16* __restrict__ A, int lda,
                                 const _Float16* __restrict__ B, int ldb,
                                 int kLen, v8f (&acc)[8]) {
  for (int k = 0; k < kLen; k += 32) {
    v16h a0 = load_a_f16(A + k, lda);
    v16h a1 = load_a_f16(A + 16 * lda + k, lda);
    v16h b0 = load_bt_f16(B + k, ldb);
    v16h b1 = load_bt_f16(B + 16 * ldb + k, ldb);
    v16h b2 = load_bt_f16(B + 32 * ldb + k, ldb);
    v16h b3 = load_bt_f16(B + 48 * ldb + k, ldb);
    acc[0] = wmma_f16(a0, b0, acc[0]);
    acc[1] = wmma_f16(a0, b1, acc[1]);
    acc[2] = wmma_f16(a0, b2, acc[2]);
    acc[3] = wmma_f16(a0, b3, acc[3]);
    acc[4] = wmma_f16(a1, b0, acc[4]);
    acc[5] = wmma_f16(a1, b1, acc[5]);
    acc[6] = wmma_f16(a1, b2, acc[6]);
    acc[7] = wmma_f16(a1, b3, acc[7]);
  }
}

// ---- Kernel 0: f32 -> f16 bulk convert ----------------------------------
__global__ __launch_bounds__(256) void cvt_kernel(
    const float* __restrict__ src, _Float16* __restrict__ dst, int n8) {
  const int idx = blockIdx.x * 256 + threadIdx.x;
  if (idx >= n8) return;
  const float* p = src + (size_t)idx * 8;
  v4f f0 = *(const v4f*)(p);
  v4f f1 = *(const v4f*)(p + 4);
  v8h h;
#pragma unroll
  for (int i = 0; i < 4; ++i) {
    h[i]     = (_Float16)f0[i];
    h[4 + i] = (_Float16)f1[i];
  }
  *(v8h*)(dst + (size_t)idx * 8) = h;
}

// ---- Kernel 1: QKV projection -------------------------------------------
// grid = (6, 32, 3); block = 256 (8 waves). Wave w: rows +=(w&3)*32,
// cols += (w>>2)*64 within the 128x128 block tile. z selects Q/K/V.
__global__ __launch_bounds__(256) void qkv_kernel(
    const _Float16* __restrict__ xh,
    const _Float16* __restrict__ Wqh, const _Float16* __restrict__ Wkh,
    const _Float16* __restrict__ Wvh,
    _Float16* __restrict__ Q, _Float16* __restrict__ Kh,
    _Float16* __restrict__ Vt) {
  const int wave = threadIdx.x >> 5;
  const int lane = threadIdx.x & 31;
  const int i0 = blockIdx.y * 128 + (wave & 3) * 32;   // token rows
  const int j0 = blockIdx.x * 128 + (wave >> 2) * 64;  // output features
  const int z  = blockIdx.z;
  const _Float16* W = (z == 0) ? Wqh : (z == 1) ? Wkh : Wvh;

  v8f acc[8] = {};
  gemm32x64(xh + (size_t)i0 * D_DIM, D_DIM, W + (size_t)j0 * D_DIM, D_DIM,
            D_DIM, acc);

  const int n  = lane & 15;
  const int mb = (lane >> 4) << 3;   // 0 or 8
  if (z == 2) {
    // transposed store: Vt[feature][token]; 8 consecutive tokens per lane
#pragma unroll
    for (int s = 0; s < 2; ++s)
#pragma unroll
      for (int t = 0; t < 4; ++t) {
        const int j = j0 + t * 16 + n;
        v8h pk;
#pragma unroll
        for (int v = 0; v < 8; ++v) pk[v] = (_Float16)acc[s * 4 + t][v];
        *(v8h*)(Vt + (size_t)j * N_TOK + i0 + s * 16 + mb) = pk;
      }
  } else {
    _Float16* O = (z == 0) ? Q : Kh;
    const float sc = (z == 0) ? SOFTMAX_SCALE : 1.0f;
#pragma unroll
    for (int s = 0; s < 2; ++s)
#pragma unroll
      for (int t = 0; t < 4; ++t)
#pragma unroll
        for (int v = 0; v < 8; ++v)
          O[(size_t)(i0 + s * 16 + mb + v) * D_DIM + j0 + t * 16 + n] =
              (_Float16)(acc[s * 4 + t][v] * sc);
  }
}

// ---- Kernel 2: S = Q K^T (lower-triangle blocks only) -------------------
__global__ __launch_bounds__(256) void scores_kernel(
    const _Float16* __restrict__ Q, const _Float16* __restrict__ Kh,
    _Float16* __restrict__ S) {
  if (blockIdx.x > blockIdx.y) return;   // strictly above diagonal
  const int wave = threadIdx.x >> 5;
  const int lane = threadIdx.x & 31;
  const int i0 = blockIdx.y * 128 + (wave & 3) * 32;
  const int j0 = blockIdx.x * 128 + (wave >> 2) * 64;

  v8f acc[8] = {};
  gemm32x64(Q + (size_t)i0 * D_DIM, D_DIM, Kh + (size_t)j0 * D_DIM, D_DIM,
            D_DIM, acc);

  const int n  = lane & 15;
  const int mb = (lane >> 4) << 3;
#pragma unroll
  for (int s = 0; s < 2; ++s)
#pragma unroll
    for (int t = 0; t < 4; ++t)
#pragma unroll
      for (int v = 0; v < 8; ++v)
        S[(size_t)(i0 + s * 16 + mb + v) * N_TOK + j0 + t * 16 + n] =
            (_Float16)acc[s * 4 + t][v];
}

// ---- Kernel 3: causal row softmax, in place S -> P ----------------------
__global__ __launch_bounds__(256) void softmax_kernel(_Float16* __restrict__ S) {
  __shared__ float red[256];
  const int i = blockIdx.x;
  const int L = i + 1;
  _Float16* row = S + (size_t)i * N_TOK;

  float sreg[16];
  float m = -1e30f;
#pragma unroll
  for (int t = 0; t < 16; ++t) {
    const int j = threadIdx.x + t * 256;
    const float s = (j < L) ? (float)row[j] : -1e30f;
    sreg[t] = s;
    m = fmaxf(m, s);
  }
  red[threadIdx.x] = m;
  __syncthreads();
  for (int o = 128; o > 0; o >>= 1) {
    if (threadIdx.x < (unsigned)o)
      red[threadIdx.x] = fmaxf(red[threadIdx.x], red[threadIdx.x + o]);
    __syncthreads();
  }
  m = red[0];
  __syncthreads();

  float sum = 0.0f;
#pragma unroll
  for (int t = 0; t < 16; ++t) {
    const int j = threadIdx.x + t * 256;
    const float e = (j < L) ? __expf(sreg[t] - m) : 0.0f;
    sreg[t] = e;
    sum += e;
  }
  red[threadIdx.x] = sum;
  __syncthreads();
  for (int o = 128; o > 0; o >>= 1) {
    if (threadIdx.x < (unsigned)o)
      red[threadIdx.x] += red[threadIdx.x + o];
    __syncthreads();
  }
  const float inv = 1.0f / red[0];

#pragma unroll
  for (int t = 0; t < 16; ++t) {
    const int j = threadIdx.x + t * 256;
    row[j] = (j < L) ? (_Float16)(sreg[t] * inv) : (_Float16)0.0f;
  }
}

// ---- Kernel 4: out = P V  (B^T rows come from Vt) -----------------------
__global__ __launch_bounds__(256) void pv_kernel(
    const _Float16* __restrict__ P, const _Float16* __restrict__ Vt,
    float* __restrict__ O) {
  const int wave = threadIdx.x >> 5;
  const int lane = threadIdx.x & 31;
  const int i0 = blockIdx.y * 128 + (wave & 3) * 32;
  const int d0 = blockIdx.x * 128 + (wave >> 2) * 64;
  const int nend = blockIdx.y * 128 + 128;   // keys beyond row block are zero

  v8f acc[8] = {};
  gemm32x64(P + (size_t)i0 * N_TOK, N_TOK, Vt + (size_t)d0 * N_TOK, N_TOK,
            nend, acc);

  const int n  = lane & 15;
  const int mb = (lane >> 4) << 3;
#pragma unroll
  for (int s = 0; s < 2; ++s)
#pragma unroll
    for (int t = 0; t < 4; ++t)
#pragma unroll
      for (int v = 0; v < 8; ++v)
        O[(size_t)(i0 + s * 16 + mb + v) * D_DIM + d0 + t * 16 + n] =
            acc[s * 4 + t][v];
}

// ---- Launch -------------------------------------------------------------
extern "C" void kernel_launch(void* const* d_in, const int* in_sizes, int n_in,
                              void* d_out, int out_size, void* d_ws,
                              size_t ws_size, hipStream_t stream) {
  const float* x  = (const float*)d_in[0];
  const float* Wq = (const float*)d_in[1];
  const float* Wk = (const float*)d_in[2];
  const float* Wv = (const float*)d_in[3];
  float* out = (float*)d_out;

  // Workspace layout (halves): xh | Wqh | Wkh | Wvh | Q | K | Vt | S(P)
  const size_t XN = (size_t)N_TOK * D_DIM;   // 3,145,728
  const size_t WN = (size_t)D_DIM * D_DIM;   //   589,824
  _Float16* ws  = (_Float16*)d_ws;
  _Float16* xh  = ws;
  _Float16* Wqh = xh + XN;
  _Float16* Wkh = Wqh + WN;
  _Float16* Wvh = Wkh + WN;
  _Float16* Q   = Wvh + WN;
  _Float16* Kh  = Q + XN;
  _Float16* Vt  = Kh + XN;
  _Float16* S   = Vt + XN;

  dim3 blk(256);
  cvt_kernel<<<dim3((int)(XN / 8 / 256)), blk, 0, stream>>>(x, xh, (int)(XN / 8));
  cvt_kernel<<<dim3((int)(WN / 8 / 256)), blk, 0, stream>>>(Wq, Wqh, (int)(WN / 8));
  cvt_kernel<<<dim3((int)(WN / 8 / 256)), blk, 0, stream>>>(Wk, Wkh, (int)(WN / 8));
  cvt_kernel<<<dim3((int)(WN / 8 / 256)), blk, 0, stream>>>(Wv, Wvh, (int)(WN / 8));

  qkv_kernel<<<dim3(D_DIM / 128, N_TOK / 128, 3), blk, 0, stream>>>(
      xh, Wqh, Wkh, Wvh, Q, Kh, Vt);
  scores_kernel<<<dim3(N_TOK / 128, N_TOK / 128), blk, 0, stream>>>(Q, Kh, S);
  softmax_kernel<<<dim3(N_TOK), blk, 0, stream>>>(S);
  pv_kernel<<<dim3(D_DIM / 128, N_TOK / 128), blk, 0, stream>>>(S, Vt, out);
}
